// FGDN_11184094839450
// MI455X (gfx1250) — matrix operand order repaired and checked by reading, not compiled, not run on hardware
//
#include <hip/hip_runtime.h>

// ---------------------------------------------------------------------------
// ChebConv GNN forward for MI455X (gfx1250, wave32, WMMA).
// Dense GEMMs: v_wmma_f32_16x16x32_bf16 with LDS-staged A fragments
// (one 512-thread block = 16 waves = full 256-wide output strip).
// Sparse props: f32 atomics (L2-resident working set).
// ---------------------------------------------------------------------------

typedef __bf16  v16bf __attribute__((ext_vector_type(16)));
typedef __bf16  v8bf  __attribute__((ext_vector_type(8)));
typedef float   v8f   __attribute__((ext_vector_type(8)));

constexpr int NN  = 10000;   // nodes
constexpr int NE  = 320000;  // edges
constexpr int FIN = 128;     // input feature dim
constexpr int FD  = 256;     // hidden dim
constexpr int NC  = 10;      // classes
constexpr int NG  = 64;      // graphs
constexpr int LDS_PAD = 8;   // bf16 elements -> 16B -> 4-bank shift per row

// ------------------------- degree / edge weights ---------------------------

__global__ void deg_kernel(const int* __restrict__ row, float* __restrict__ deg, int E) {
    int e = blockIdx.x * blockDim.x + threadIdx.x;
    if (e < E) atomicAdd(&deg[row[e]], 1.0f);
}

__global__ void dinv_kernel(const float* __restrict__ deg, float* __restrict__ dinv, int n) {
    int i = blockIdx.x * blockDim.x + threadIdx.x;
    if (i < n) {
        float d = deg[i];
        dinv[i] = (d > 0.0f) ? rsqrtf(d) : 0.0f;
    }
}

__global__ void edgew_kernel(const int* __restrict__ row, const int* __restrict__ col,
                             const float* __restrict__ dinv, float* __restrict__ w, int E) {
    int e = blockIdx.x * blockDim.x + threadIdx.x;
    if (e < E) w[e] = -dinv[row[e]] * dinv[col[e]];
}

// ------------------------- sparse propagation ------------------------------
// hout[col[e], :] += scale * w[e] * hin[row[e], :]

__global__ void prop_kernel(const float* __restrict__ hin, const int* __restrict__ row,
                            const int* __restrict__ col, const float* __restrict__ w,
                            float scale, int D4, int E, float* __restrict__ hout) {
    long long t = (long long)blockIdx.x * blockDim.x + threadIdx.x;
    long long total = (long long)E * D4;
    if (t >= total) return;
    int e = (int)(t / D4);
    int c = ((int)(t % D4)) << 2;
    int D = D4 << 2;
    float we = scale * w[e];
    const float4 v = *(const float4*)(hin + (size_t)row[e] * D + c);
    float* o = hout + (size_t)col[e] * D + c;
    atomicAdd(o + 0, we * v.x);
    atomicAdd(o + 1, we * v.y);
    atomicAdd(o + 2, we * v.z);
    atomicAdd(o + 3, we * v.w);
}

// dst = -src (seed for T2 = 2*prop(T1) - T0)
__global__ void negcopy_kernel(const float* __restrict__ src, float* __restrict__ dst,
                               long long n) {
    long long i = (long long)blockIdx.x * blockDim.x + threadIdx.x;
    if (i < n) dst[i] = -src[i];
}

// ------------------------- weight fragment packing -------------------------
// Source W: row-major [Ktot, 256] fp32 (Cheb order k0;k1;k2 stacked on K).
// Dest: per (ktile, ntile) fragment, 32 lanes x 16 contiguous bf16:
//   lane n = lane&15, K = ktile*32 + (lane>>4)*16 + i  (i = element index)

__global__ void pack_w_kernel(const float* __restrict__ W, int Ktot, __bf16* __restrict__ dst) {
    int t = blockIdx.x * blockDim.x + threadIdx.x;
    int ktiles = Ktot >> 5;
    int total = ktiles * 16 * 32;
    if (t >= total) return;
    int lane  = t & 31;
    int frag  = t >> 5;
    int ntile = frag & 15;
    int n     = ntile * 16 + (lane & 15);
    int kbase = (frag >> 4) * 32 + (lane >> 4) * 16;
    __bf16* d = dst + (size_t)t * 16;
#pragma unroll
    for (int i = 0; i < 16; ++i)
        d[i] = (__bf16)W[(size_t)(kbase + i) * 256 + n];
}

// ------------------------- WMMA ChebConv GEMM ------------------------------
// OUT[10000,256] = T0@W0 + T1@W1 + T2@W2 + bias, then PReLU(act=0)/ReLU(act=1).
// Block = 512 thr (16 waves): wave w computes N-tile w of one 16-row strip.
// A strips of T0/T1/T2 staged once into LDS as bf16 (padded, bank-conflict free).

__device__ __forceinline__ void stage_T(const float* __restrict__ T, __bf16* __restrict__ plane,
                                        int Din, int stride, int mtile, int tid) {
    int ept = (16 * Din) >> 9;          // elements per thread (512 threads): 4 or 8
    int base = tid * ept;
    int m = base / Din;
    int k = base - m * Din;
    const float* src = T + (size_t)(mtile * 16 + m) * Din + k;
    __bf16* dst = plane + m * stride + k;
#pragma unroll 8
    for (int i = 0; i < ept; ++i) dst[i] = (__bf16)src[i];
}

__device__ __forceinline__ void acc_one_T(v8f& c, const __bf16* __restrict__ plane,
                                          int stride, int lane,
                                          const __bf16* __restrict__ Wp,
                                          int tIdx, int ktiles, int ntile) {
    int m    = lane & 15;
    int half = lane >> 4;
    for (int kt = 0; kt < ktiles; ++kt) {
        const __bf16* ap = plane + m * stride + kt * 32 + half * 16;
        v8bf lo = *(const v8bf*)ap;          // K = kk + half*16 + [0..7]
        v8bf hi = *(const v8bf*)(ap + 8);    // K = kk + half*16 + 8 + [0..7]
        v16bf a;
#pragma unroll
        for (int i = 0; i < 8; ++i) { a[i] = lo[i]; a[8 + i] = hi[i]; }
        const v16bf b = *(const v16bf*)(Wp +
            (((size_t)((tIdx * ktiles + kt) * 16 + ntile)) * 32 + lane) * 16);
        c = __builtin_amdgcn_wmma_f32_16x16x32_bf16(
                /*neg_a=*/false, a, /*neg_b=*/false, b,
                /*c_mod=*/(short)0, c, /*reuse_a=*/false, /*reuse_b=*/false);
    }
}

__global__ void __launch_bounds__(512)
gemm_cheb_kernel(const float* __restrict__ T0, const float* __restrict__ T1,
                 const float* __restrict__ T2, int Din,
                 const __bf16* __restrict__ Wp, const float* __restrict__ bias,
                 int act, const float* __restrict__ alpha, float* __restrict__ out) {
    __shared__ __bf16 smem[3 * 16 * (FD + LDS_PAD)];

    int tid   = threadIdx.x;
    int mtile = blockIdx.x;
    int stride = Din + LDS_PAD;
    __bf16* p0 = smem;
    __bf16* p1 = smem + 16 * stride;
    __bf16* p2 = smem + 32 * stride;

    stage_T(T0, p0, Din, stride, mtile, tid);
    stage_T(T1, p1, Din, stride, mtile, tid);
    stage_T(T2, p2, Din, stride, mtile, tid);
    __syncthreads();

    int lane  = tid & 31;
    int ntile = tid >> 5;            // 16 waves -> 16 N-tiles
    int ktiles = Din >> 5;

    v8f c = {};
    acc_one_T(c, p0, stride, lane, Wp, 0, ktiles, ntile);
    acc_one_T(c, p1, stride, lane, Wp, 1, ktiles, ntile);
    acc_one_T(c, p2, stride, lane, Wp, 2, ktiles, ntile);

    int n  = ntile * 16 + (lane & 15);
    float bv = bias[n];
    float al = alpha ? alpha[0] : 0.0f;
    int mbase = mtile * 16 + ((lane >> 4) << 3);
#pragma unroll
    for (int r = 0; r < 8; ++r) {
        float v = c[r] + bv;
        v = (act == 0) ? (v >= 0.0f ? v : al * v) : fmaxf(v, 0.0f);
        out[(size_t)(mbase + r) * 256 + n] = v;
    }
}

// ------------------------- global add pool ---------------------------------

__global__ void pool_kernel(const float* __restrict__ h, const int* __restrict__ batch,
                            float* __restrict__ g) {
    int t = blockIdx.x * blockDim.x + threadIdx.x;   // NN * 64 threads (float4 cols)
    if (t >= NN * (FD / 4)) return;
    int i = t / (FD / 4);
    int c = (t % (FD / 4)) << 2;
    int b = batch[i];
    const float4 v = *(const float4*)(h + (size_t)i * FD + c);
    float* o = g + (size_t)b * FD + c;
    atomicAdd(o + 0, v.x);
    atomicAdd(o + 1, v.y);
    atomicAdd(o + 2, v.z);
    atomicAdd(o + 3, v.w);
}

// ------------------------- MLP head ----------------------------------------

__global__ void fc1_kernel(const float* __restrict__ g, const float* __restrict__ w,
                           const float* __restrict__ b, const float* __restrict__ alpha,
                           float* __restrict__ z) {
    int idx = blockIdx.x * blockDim.x + threadIdx.x;   // NG*256
    if (idx >= NG * FD) return;
    int gi = idx >> 8, j = idx & 255;
    float acc = b[j];
    for (int k = 0; k < FD; ++k) acc += g[gi * FD + k] * w[k * FD + j];
    float al = alpha[0];
    z[idx] = (acc >= 0.0f) ? acc : al * acc;
}

__global__ void fc2_kernel(const float* __restrict__ z1, const float* __restrict__ w,
                           const float* __restrict__ b, float* __restrict__ z2) {
    int idx = blockIdx.x * blockDim.x + threadIdx.x;   // NG*128
    if (idx >= NG * (FD / 2)) return;
    int gi = idx / (FD / 2), j = idx % (FD / 2);
    float acc = b[j];
    for (int k = 0; k < FD; ++k) acc += z1[gi * FD + k] * w[k * (FD / 2) + j];
    z2[idx] = 1.0f / (1.0f + __expf(-acc));
}

__global__ void fc3_lsm_kernel(const float* __restrict__ z2, const float* __restrict__ w,
                               const float* __restrict__ b, float* __restrict__ out) {
    int gi = blockIdx.x * blockDim.x + threadIdx.x;
    if (gi >= NG) return;
    float logit[NC];
    float mx = -1e30f;
#pragma unroll
    for (int j = 0; j < NC; ++j) {
        float acc = b[j];
        for (int k = 0; k < FD / 2; ++k) acc += z2[gi * (FD / 2) + k] * w[k * NC + j];
        logit[j] = acc;
        mx = fmaxf(mx, acc);
    }
    float s = 0.0f;
#pragma unroll
    for (int j = 0; j < NC; ++j) s += __expf(logit[j] - mx);
    float ls = __logf(s) + mx;
#pragma unroll
    for (int j = 0; j < NC; ++j) out[gi * NC + j] = logit[j] - ls;
}

// ---------------------------------------------------------------------------

extern "C" void kernel_launch(void* const* d_in, const int* in_sizes, int n_in,
                              void* d_out, int out_size, void* d_ws, size_t ws_size,
                              hipStream_t stream) {
    (void)in_sizes; (void)n_in; (void)out_size; (void)ws_size;

    const float* x     = (const float*)d_in[0];        // [NN, FIN]
    const int*   eidx  = (const int*)d_in[1];          // [2, NE]
    const int*   batch = (const int*)d_in[2];          // [NN]
    const float* W1    = (const float*)d_in[3];        // [3, FIN, FD] == [384, 256]
    const float* b1    = (const float*)d_in[4];        // [FD]
    const float* Wr    = (const float*)d_in[5];        // [3, 3, FD, FD]
    const float* br    = (const float*)d_in[6];        // [3, FD]
    const float* fc1w  = (const float*)d_in[7];
    const float* fc1b  = (const float*)d_in[8];
    const float* fc2w  = (const float*)d_in[9];
    const float* fc2b  = (const float*)d_in[10];
    const float* fc3w  = (const float*)d_in[11];
    const float* fc3b  = (const float*)d_in[12];
    const float* a1    = (const float*)d_in[13];
    const float* a3    = (const float*)d_in[14];

    const int* row = eidx;
    const int* col = eidx + NE;

    // ---- workspace carve-up ----
    char* ws = (char*)d_ws;
    size_t off = 0;
    auto alloc = [&](size_t bytes) {
        size_t o = off;
        off += (bytes + 255) & ~(size_t)255;
        return o;
    };
    float* deg   = (float*)(ws + alloc((size_t)NN * 4));
    float* dinv  = (float*)(ws + alloc((size_t)NN * 4));
    float* ew    = (float*)(ws + alloc((size_t)NE * 4));
    float* T1    = (float*)(ws + alloc((size_t)NN * FD * 4));
    float* T2    = (float*)(ws + alloc((size_t)NN * FD * 4));
    float* H0    = (float*)(ws + alloc((size_t)NN * FD * 4));
    float* H1    = (float*)(ws + alloc((size_t)NN * FD * 4));
    const size_t wp1_elems = (size_t)(3 * FIN / 32) * 16 * 32 * 16;   // layer-1 frags
    const size_t wpr_elems = (size_t)(3 * FD / 32) * 16 * 32 * 16;    // hidden frags
    __bf16* WpL1 = (__bf16*)(ws + alloc(wp1_elems * 2));
    __bf16* WpR0 = (__bf16*)(ws + alloc(wpr_elems * 2));
    __bf16* WpR1 = (__bf16*)(ws + alloc(wpr_elems * 2));
    __bf16* WpR2 = (__bf16*)(ws + alloc(wpr_elems * 2));
    float* gpool = (float*)(ws + alloc((size_t)NG * FD * 4));
    float* z1    = (float*)(ws + alloc((size_t)NG * FD * 4));
    float* z2    = (float*)(ws + alloc((size_t)NG * (FD / 2) * 4));

    const int TB = 256;
    auto blocks = [](long long n, int tb) { return (unsigned)((n + tb - 1) / tb); };

    // ---- edge weights ----
    hipMemsetAsync(deg, 0, (size_t)NN * 4, stream);
    deg_kernel<<<blocks(NE, TB), TB, 0, stream>>>(row, deg, NE);
    dinv_kernel<<<blocks(NN, TB), TB, 0, stream>>>(deg, dinv, NN);
    edgew_kernel<<<blocks(NE, TB), TB, 0, stream>>>(row, col, dinv, ew, NE);

    // ---- pack weights into B-fragment layout (bf16) ----
    pack_w_kernel<<<blocks((long long)(3 * FIN / 32) * 16 * 32, TB), TB, 0, stream>>>(W1, 3 * FIN, WpL1);
    pack_w_kernel<<<blocks((long long)(3 * FD / 32) * 16 * 32, TB), TB, 0, stream>>>(Wr + 0 * 3 * FD * FD, 3 * FD, WpR0);
    pack_w_kernel<<<blocks((long long)(3 * FD / 32) * 16 * 32, TB), TB, 0, stream>>>(Wr + 1 * 3 * FD * FD, 3 * FD, WpR1);
    pack_w_kernel<<<blocks((long long)(3 * FD / 32) * 16 * 32, TB), TB, 0, stream>>>(Wr + 2 * 3 * FD * FD, 3 * FD, WpR2);

    // ---- layer 1 (Din = 128, PReLU) ----
    {
        const int Din = FIN, D4 = Din / 4;
        hipMemsetAsync(T1, 0, (size_t)NN * Din * 4, stream);
        prop_kernel<<<blocks((long long)NE * D4, TB), TB, 0, stream>>>(x, row, col, ew, 1.0f, D4, NE, T1);
        negcopy_kernel<<<blocks((long long)NN * Din, TB), TB, 0, stream>>>(x, T2, (long long)NN * Din);
        prop_kernel<<<blocks((long long)NE * D4, TB), TB, 0, stream>>>(T1, row, col, ew, 2.0f, D4, NE, T2);
        gemm_cheb_kernel<<<NN / 16, 512, 0, stream>>>(x, T1, T2, Din, WpL1, b1,
                                                      /*act=*/0, a1, H0);
    }

    // ---- hidden layers (Din = 256, ReLU) ----
    __bf16* WpR[3] = {WpR0, WpR1, WpR2};
    float* Hprev = H0;
    float* Hnext = H1;
    for (int i = 0; i < 3; ++i) {
        const int Din = FD, D4 = Din / 4;
        hipMemsetAsync(T1, 0, (size_t)NN * Din * 4, stream);
        prop_kernel<<<blocks((long long)NE * D4, TB), TB, 0, stream>>>(Hprev, row, col, ew, 1.0f, D4, NE, T1);
        negcopy_kernel<<<blocks((long long)NN * Din, TB), TB, 0, stream>>>(Hprev, T2, (long long)NN * Din);
        prop_kernel<<<blocks((long long)NE * D4, TB), TB, 0, stream>>>(T1, row, col, ew, 2.0f, D4, NE, T2);
        gemm_cheb_kernel<<<NN / 16, 512, 0, stream>>>(Hprev, T1, T2, Din, WpR[i], br + i * FD,
                                                      /*act=*/1, nullptr, Hnext);
        float* t = Hprev; Hprev = Hnext; Hnext = t;
    }

    // ---- global add pool ----
    hipMemsetAsync(gpool, 0, (size_t)NG * FD * 4, stream);
    pool_kernel<<<blocks((long long)NN * (FD / 4), TB), TB, 0, stream>>>(Hprev, batch, gpool);

    // ---- MLP head ----
    fc1_kernel<<<blocks((long long)NG * FD, TB), TB, 0, stream>>>(gpool, fc1w, fc1b, a3, z1);
    fc2_kernel<<<blocks((long long)NG * (FD / 2), TB), TB, 0, stream>>>(z1, fc2w, fc2b, z2);
    fc3_lsm_kernel<<<1, 64, 0, stream>>>(z2, fc3w, fc3b, (float*)d_out);
}
